// MySelfAttention_24790551232636
// MI455X (gfx1250) — matrix operand order repaired and checked by reading, not compiled, not run on hardware
//
#include <hip/hip_runtime.h>

typedef _Float16 v16h __attribute__((ext_vector_type(16)));
typedef _Float16 v8h  __attribute__((ext_vector_type(8)));
typedef _Float16 v4h  __attribute__((ext_vector_type(4)));
typedef float    v8f  __attribute__((ext_vector_type(8)));

#define BATCH  2
#define SEQ    2048
#define HEADS  16
#define DHEAD  64
#define DMODEL 1024
#define MROWS  (BATCH * SEQ)   // 4096

static __device__ __forceinline__ v16h pack16(v8h lo, v8h hi) {
  v16h r;
#pragma unroll
  for (int i = 0; i < 8; ++i) { r[i] = lo[i]; r[i + 8] = hi[i]; }
  return r;
}

static __device__ __forceinline__ v8f wmma_f16(v16h a, v16h b, v8f c) {
  // D(16x16,f32) = A(16x32,f16) * B(32x16,f16) + C
  return __builtin_amdgcn_wmma_f32_16x16x32_f16(false, a, false, b, (short)0, c,
                                                false, false);
}

// Async global -> LDS DMA (ASYNCcnt-tracked), 16 bytes per enabled lane.
static __device__ __forceinline__ void async_ld_b128(void* lds_ptr,
                                                     const void* gptr) {
  unsigned int lds_off = (unsigned int)(unsigned long long)lds_ptr;
  asm volatile("global_load_async_to_lds_b128 %0, %1, off"
               :: "v"(lds_off), "v"(gptr)
               : "memory");
}

// ---------------------------------------------------------------------------
// Kernel 1: X fp32 -> f16 (4 elements / thread)
// ---------------------------------------------------------------------------
__global__ __launch_bounds__(256) void cvt_x_kernel(const float* __restrict__ x,
                                                    _Float16* __restrict__ xh) {
  int i = blockIdx.x * 256 + threadIdx.x;           // i < MROWS*DMODEL/4
  float4 v = ((const float4*)x)[i];
  v4h o;
  o[0] = (_Float16)v.x; o[1] = (_Float16)v.y;
  o[2] = (_Float16)v.z; o[3] = (_Float16)v.w;
  ((v4h*)xh)[i] = o;
}

// ---------------------------------------------------------------------------
// Kernel 2: weights fp32 [K][N] -> f16 transposed [N][K] (for WMMA B frags)
// ---------------------------------------------------------------------------
__global__ __launch_bounds__(256) void cvt_w_kernel(
    const float* __restrict__ Wq, const float* __restrict__ Wk,
    const float* __restrict__ Wv, _Float16* __restrict__ Wqt,
    _Float16* __restrict__ Wkt, _Float16* __restrict__ Wvt) {
  int i = blockIdx.x * 256 + threadIdx.x;           // i < 3*1024*1024
  int which = i >> 20;
  int r = i & 0xFFFFF;
  int k = r >> 10;
  int n = r & 1023;
  const float* src = (which == 0) ? Wq : ((which == 1) ? Wk : Wv);
  _Float16* dst = (which == 0) ? Wqt : ((which == 1) ? Wkt : Wvt);
  dst[(size_t)n * DMODEL + k] = (_Float16)src[r];
}

// ---------------------------------------------------------------------------
// Kernel 3: QKV projection GEMM. One wave -> 32x64 output tile.
// The 8 waves of a block share the same 64 weight columns: the 64x32 weight
// tile is staged once per k-step into double-buffered LDS with async
// global->LDS DMA (one 16B DMA per thread). B fragments are read from LDS
// as a clause of 8 ds_load_b128 into distinct registers, then 8 WMMAs run,
// so LDS latency overlaps XDL execution. A fragments are register
// double-buffered (explicit ping-pong, no rotation copies).
//   p=0: Q[B,H,S,64]   p=1: K[B,H,S,64]   p=2: V transposed [B,H,64,S]
// ---------------------------------------------------------------------------
__global__ __launch_bounds__(256) void qkv_gemm_kernel(
    const _Float16* __restrict__ Xh, const _Float16* __restrict__ Wqt,
    const _Float16* __restrict__ Wkt, const _Float16* __restrict__ Wvt,
    const float* __restrict__ bq, const float* __restrict__ bk,
    const float* __restrict__ bv, _Float16* __restrict__ Q,
    _Float16* __restrict__ K, _Float16* __restrict__ VT) {
  __shared__ _Float16 wtile[2][64 * 32];  // 64 weight cols x 32 k, 2 x 4KB
  const int tid = threadIdx.x;
  const int lane = tid & 31;
  const int gw = blockIdx.x * 8 + (tid >> 5);
  const int p = gw >> 11;                 // 0..2 (2048 tiles per projection)
  const int rem = gw & 2047;
  const int nt = rem >> 7;                // 0..15, same for all waves in block
  const int mt = rem & 127;               // 0..127 (32-row tiles)
  const int m0 = mt << 5;
  const int n0 = nt << 6;
  const int lm = lane & 15, lk = lane >> 4;

  const _Float16* __restrict__ W = (p == 0) ? Wqt : ((p == 1) ? Wkt : Wvt);
  const float* __restrict__ bias = (p == 0) ? bq : ((p == 1) ? bk : bv);

  // cooperative staging: thread moves 16B of the 64x32 W tile per k-step
  const int wrow_s = tid >> 2;            // 0..63 -> weight column n0+wrow_s
  const int wcol_s = (tid & 3) * 8;       // k-chunk within the 32-k tile
  const _Float16* wsrc = W + (size_t)(n0 + wrow_s) * DMODEL + wcol_s;

  const _Float16* a0lo = Xh + (size_t)(m0 + lm) * DMODEL + 16 * lk;
  const _Float16* a0hi = a0lo + 8;
  const _Float16* a1lo = Xh + (size_t)(m0 + 16 + lm) * DMODEL + 16 * lk;
  const _Float16* a1hi = a1lo + 8;

  v8f acc[2][4] = {};
  v16h A0a, A0b, A1a, A1b;                // two ping-pong A-fragment sets

  auto stage = [&](int buf, int k) {
    async_ld_b128(&wtile[buf][wrow_s * 32 + wcol_s], wsrc + k);
  };
  auto loadA0 = [&](int k) {
    A0a = pack16(*(const v8h*)(a0lo + k), *(const v8h*)(a0hi + k));
    A0b = pack16(*(const v8h*)(a1lo + k), *(const v8h*)(a1hi + k));
  };
  auto loadA1 = [&](int k) {
    A1a = pack16(*(const v8h*)(a0lo + k), *(const v8h*)(a0hi + k));
    A1b = pack16(*(const v8h*)(a1lo + k), *(const v8h*)(a1hi + k));
  };
  auto mmas = [&](v16h aa, v16h ab, const _Float16* bt) {
    v16h bf[4];
#pragma unroll
    for (int j = 0; j < 4; ++j)   // clause of LDS loads, distinct registers
      bf[j] = *(const v16h*)(bt + (16 * j + lm) * 32 + 16 * lk);
#pragma unroll
    for (int j = 0; j < 4; ++j) {
      acc[0][j] = wmma_f16(aa, bf[j], acc[0][j]);
      acc[1][j] = wmma_f16(ab, bf[j], acc[1][j]);
    }
  };

  stage(0, 0);
  loadA0(0);
  for (int it = 0; it < 32; it += 2) {    // 16 double iterations, 32 k-steps
    const int k0 = it * 32;
    stage(1, k0 + 32);                    // k0+32 <= 992: always valid
    loadA1(k0 + 32);
    asm volatile("s_wait_asynccnt 1" ::: "memory");  // buf0 tile complete
    __syncthreads();
    mmas(A0a, A0b, &wtile[0][0]);
    __syncthreads();
    if (it + 2 < 32) {
      stage(0, k0 + 64);
      loadA0(k0 + 64);
      asm volatile("s_wait_asynccnt 1" ::: "memory");  // buf1 tile complete
    } else {
      asm volatile("s_wait_asynccnt 0" ::: "memory");
    }
    __syncthreads();
    mmas(A1a, A1b, &wtile[1][0]);
    __syncthreads();
  }

#pragma unroll
  for (int j = 0; j < 4; ++j) {
    const int n = n0 + 16 * j + lm;       // this lane's output column
    const float bb = bias[n];
    const int h = n >> 6, d = n & 63;
#pragma unroll
    for (int g = 0; g < 2; ++g)
#pragma unroll
      for (int r = 0; r < 8; ++r) {       // C layout: VGPR r -> row r + 8*lk
        const int m = m0 + 16 * g + r + 8 * lk;
        const int b = m >> 11, s = m & 2047;
        const _Float16 val = (_Float16)(acc[g][j][r] + bb);
        if (p == 0)
          Q[(((size_t)(b * HEADS + h)) * SEQ + s) * DHEAD + d] = val;
        else if (p == 1)
          K[(((size_t)(b * HEADS + h)) * SEQ + s) * DHEAD + d] = val;
        else
          VT[(((size_t)(b * HEADS + h)) * DHEAD + d) * SEQ + s] = val;
      }
  }
}

// ---------------------------------------------------------------------------
// Kernel 4: flash attention. One wave -> one (b,h,16-query) tile; the 8
// waves of a block share (b,h) and cooperatively stage the 32-key K and V
// tiles into double-buffered LDS with async global->LDS DMA. V fragments
// are loaded right after the score WMMAs so their LDS latency hides under
// the softmax VALU work.
// ---------------------------------------------------------------------------
__global__ __launch_bounds__(256) void attn_kernel(
    const _Float16* __restrict__ Q, const _Float16* __restrict__ K,
    const _Float16* __restrict__ VT, const float* __restrict__ mask,
    float* __restrict__ out) {
  __shared__ _Float16 ktile[2][32 * DHEAD];   // keys x dims, 2 x 4KB
  __shared__ _Float16 vtile[2][DHEAD * 32];   // dims x keys, 2 x 4KB
  __shared__ _Float16 ldsP[8][16 * 32];       // per-wave P staging, 8KB
  const int tid = threadIdx.x;
  const int lane = tid & 31;
  const int wv = tid >> 5;
  const int gw = blockIdx.x * 8 + wv;
  const int qt = gw & 127;                // 128 query tiles per (b,h)
  const int bh = gw >> 7;                 // same for all 8 waves in block
  const int b = bh >> 4;
  const int h = bh & 15;
  const int q0 = qt << 4;
  const int lm = lane & 15, lk = lane >> 4;

  const _Float16* __restrict__ Qbh = Q + (size_t)bh * SEQ * DHEAD;
  const _Float16* __restrict__ Kbh = K + (size_t)bh * SEQ * DHEAD;
  const _Float16* __restrict__ Vbh = VT + (size_t)bh * DHEAD * SEQ;
  const float* __restrict__ mrow = mask + (size_t)b * SEQ;

  // cooperative-staging addresses for this thread (16B K chunk, 16B V chunk)
  const int krow_s = tid >> 3, kcol_s = (tid & 7) * 8;    // K: 32 x 64 halves
  const int vrow_s = tid >> 2, vcol_s = (tid & 3) * 8;    // V: 64 x 32 halves

  // Q A-fragments for head-dims [0,32) and [32,64)
  const _Float16* qrow = Qbh + (size_t)(q0 + lm) * DHEAD;
  v16h aQ0 = pack16(*(const v8h*)(qrow + 16 * lk),
                    *(const v8h*)(qrow + 8 + 16 * lk));
  v16h aQ1 = pack16(*(const v8h*)(qrow + 32 + 16 * lk),
                    *(const v8h*)(qrow + 40 + 16 * lk));

  float rmax[8], rsum[8];
#pragma unroll
  for (int r = 0; r < 8; ++r) { rmax[r] = -1.0e30f; rsum[r] = 0.0f; }
  v8f accO[4] = {};

  _Float16* P = &ldsP[wv][0];
  const float scl = 0.125f;               // 1/sqrt(64)

  // prologue: stage tile 0
  async_ld_b128(&ktile[0][krow_s * DHEAD + kcol_s],
                Kbh + (size_t)krow_s * DHEAD + kcol_s);
  async_ld_b128(&vtile[0][vrow_s * 32 + vcol_s],
                Vbh + (size_t)vrow_s * SEQ + vcol_s);

  for (int it = 0; it < SEQ / 32; ++it) {
    const int k0 = it * 32;
    const int cur = it & 1;
    if (it + 1 < SEQ / 32) {
      // stage next tile into other buffer, then wait for *current* tile
      const int kn = k0 + 32;
      async_ld_b128(&ktile[cur ^ 1][krow_s * DHEAD + kcol_s],
                    Kbh + (size_t)(kn + krow_s) * DHEAD + kcol_s);
      async_ld_b128(&vtile[cur ^ 1][vrow_s * 32 + vcol_s],
                    Vbh + (size_t)vrow_s * SEQ + kn + vcol_s);
      asm volatile("s_wait_asynccnt 2" ::: "memory");
    } else {
      asm volatile("s_wait_asynccnt 0" ::: "memory");
    }
    __syncthreads();                      // current buffer ready block-wide
    const _Float16* kt = &ktile[cur][0];
    const _Float16* vt = &vtile[cur][0];

    // ---- scores: load all K fragments, then 4 WMMAs ---------------------
    v16h bk[2][2];
#pragma unroll
    for (int t = 0; t < 2; ++t) {
      const _Float16* krow = kt + (16 * t + lm) * DHEAD + 16 * lk;
      bk[t][0] = *(const v16h*)(krow);        // dims 0..31
      bk[t][1] = *(const v16h*)(krow + 32);   // dims 32..63
    }
    v8f cs[2];
#pragma unroll
    for (int t = 0; t < 2; ++t) {
      v8f c = {};
      c = wmma_f16(aQ0, bk[t][0], c);
      c = wmma_f16(aQ1, bk[t][1], c);
      cs[t] = c;
    }
    // V fragments for this tile: issue now so LDS latency hides in softmax
    v16h bv4[4];
#pragma unroll
    for (int j = 0; j < 4; ++j)
      bv4[j] = *(const v16h*)(vt + (16 * j + lm) * 32 + 16 * lk);

    const float mk0 = mrow[k0 + lm];
    const float mk1 = mrow[k0 + 16 + lm];
#pragma unroll
    for (int r = 0; r < 8; ++r) {
      cs[0][r] = cs[0][r] * scl + mk0;
      cs[1][r] = cs[1][r] * scl + mk1;
    }
    // ---- online softmax (row r+8*lk lives across a 16-lane half) --------
#pragma unroll
    for (int r = 0; r < 8; ++r) {
      float tm = fmaxf(cs[0][r], cs[1][r]);
#pragma unroll
      for (int off = 1; off < 16; off <<= 1)
        tm = fmaxf(tm, __shfl_xor(tm, off, 32));
      const float nm = fmaxf(rmax[r], tm);
      const float sc = __expf(rmax[r] - nm);
      const float p0 = __expf(cs[0][r] - nm);
      const float p1 = __expf(cs[1][r] - nm);
      float ts = p0 + p1;
#pragma unroll
      for (int off = 1; off < 16; off <<= 1)
        ts += __shfl_xor(ts, off, 32);
      rsum[r] = rsum[r] * sc + ts;
      rmax[r] = nm;
      cs[0][r] = p0;
      cs[1][r] = p1;
#pragma unroll
      for (int j = 0; j < 4; ++j) accO[j][r] *= sc;
    }
    // ---- C-layout -> per-wave LDS P tile (row-major 16x32 f16) ----------
#pragma unroll
    for (int t = 0; t < 2; ++t)
#pragma unroll
      for (int r = 0; r < 8; ++r)
        P[(r + 8 * lk) * 32 + 16 * t + lm] = (_Float16)cs[t][r];
    // wave-private tile; DS ops in-order within a wave -> no barrier
    v16h aP = pack16(*(const v8h*)(P + lm * 32 + 16 * lk),
                     *(const v8h*)(P + lm * 32 + 8 + 16 * lk));
    // ---- P @ V ----------------------------------------------------------
#pragma unroll
    for (int j = 0; j < 4; ++j) accO[j] = wmma_f16(aP, bv4[j], accO[j]);
    __syncthreads();   // all waves done with buffer `cur` before next DMA
  }

  // ---- epilogue: out[b][s][h*64 + d], fp32 ------------------------------
#pragma unroll
  for (int j = 0; j < 4; ++j) {
    const int d = h * DHEAD + 16 * j + lm;
#pragma unroll
    for (int r = 0; r < 8; ++r) {
      const int s = q0 + r + 8 * lk;
      out[((size_t)b * SEQ + s) * DMODEL + d] = accO[j][r] / rsum[r];
    }
  }
}

// ---------------------------------------------------------------------------
extern "C" void kernel_launch(void* const* d_in, const int* in_sizes, int n_in,
                              void* d_out, int out_size, void* d_ws,
                              size_t ws_size, hipStream_t stream) {
  (void)in_sizes; (void)n_in; (void)out_size; (void)ws_size;
  const float* X   = (const float*)d_in[0];
  const float* msk = (const float*)d_in[1];
  const float* Wq  = (const float*)d_in[2];
  const float* bq  = (const float*)d_in[3];
  const float* Wk  = (const float*)d_in[4];
  const float* bk  = (const float*)d_in[5];
  const float* Wv  = (const float*)d_in[6];
  const float* bv  = (const float*)d_in[7];
  float* out = (float*)d_out;

  char* w = (char*)d_ws;
  _Float16* Xh  = (_Float16*)w; w += (size_t)MROWS * DMODEL * 2;   // 8 MiB
  _Float16* Wqt = (_Float16*)w; w += (size_t)DMODEL * DMODEL * 2;  // 2 MiB
  _Float16* Wkt = (_Float16*)w; w += (size_t)DMODEL * DMODEL * 2;
  _Float16* Wvt = (_Float16*)w; w += (size_t)DMODEL * DMODEL * 2;
  _Float16* Qh  = (_Float16*)w; w += (size_t)MROWS * DMODEL * 2;   // 8 MiB
  _Float16* Kh  = (_Float16*)w; w += (size_t)MROWS * DMODEL * 2;
  _Float16* VTh = (_Float16*)w; w += (size_t)MROWS * DMODEL * 2;

  cvt_x_kernel<<<(MROWS * DMODEL / 4) / 256, 256, 0, stream>>>(X, Xh);
  cvt_w_kernel<<<(3 * DMODEL * DMODEL) / 256, 256, 0, stream>>>(
      Wq, Wk, Wv, Wqt, Wkt, Wvt);
  // 3 projections * 128 M-tiles(32 rows) * 16 N-groups = 6144 waves
  qkv_gemm_kernel<<<6144 / 8, 256, 0, stream>>>(Xh, Wqt, Wkt, Wvt, bq, bk, bv,
                                                Qh, Kh, VTh);
  // 2*16*128 = 4096 waves, 8 waves/block
  attn_kernel<<<4096 / 8, 256, 0, stream>>>(Qh, Kh, VTh, msk, out);
}
